// MultiHeadedSelfAttention_71116068487378
// MI455X (gfx1250) — compile-verified, matrix operand order checked by reading, not run on hardware
//
#include <hip/hip_runtime.h>

#define BQ 4
#define NQ 2048
#define CQ 1024
#define HQ 16
#define DQ 64
// SCALE = D^-0.5 = 0.125

// Padded LDS row strides (elements). 144 B / 80 B row pitches rotate the 64x4B
// banks so 16 lanes x 16 B fragment reads hit distinct bank windows.
#define LDK 72   // for 64-wide tiles  (144 B pitch)
#define LDV 40   // for 32-wide tiles  ( 80 B pitch)

typedef _Float16 v16h __attribute__((ext_vector_type(16)));
typedef _Float16 v8h  __attribute__((ext_vector_type(8)));
typedef float    v8f  __attribute__((ext_vector_type(8)));

static __device__ __forceinline__ v8f wmma_f16(v16h a, v16h b, v8f c) {
    // (neg_a, A, neg_b, B, c_mod, C, reuse_a, reuse_b)
    return __builtin_amdgcn_wmma_f32_16x16x32_f16(false, a, false, b, (short)0, c, false, false);
}

// Load one 16x32 f16 fragment (A pattern; also used for B via row-major "B^T" storage).
// ISA 16-bit A layout: lane L holds row M=L%16; elements 0..7 = K = 8*(L/16)+0..7,
// elements 8..15 = K = 16 + 8*(L/16) + 0..7.
static __device__ __forceinline__ v16h load_frag16(const _Float16* p, int ld, int lane) {
    const int row  = lane & 15;
    const int half = (lane >> 4) & 1;
    const _Float16* q = p + (size_t)row * (size_t)ld + half * 8;
    v8h lo = *(const v8h*)(q);
    v8h hi = *(const v8h*)(q + 16);
    return __builtin_shufflevector(lo, hi, 0, 1, 2, 3, 4, 5, 6, 7,
                                           8, 9, 10, 11, 12, 13, 14, 15);
}

// Async bulk copy: 16 bytes per lane, global -> LDS, tracked by ASYNCcnt.
static __device__ __forceinline__ void async_copy_b128(unsigned lds_off, const void* gaddr) {
    asm volatile("global_load_async_to_lds_b128 %0, %1, off"
                 :: "v"(lds_off), "v"(gaddr) : "memory");
}

// ---------------- conversion kernels ----------------

__global__ void cvt_f32_to_f16(const float* __restrict__ src, _Float16* __restrict__ dst, int n) {
    int i = blockIdx.x * blockDim.x + threadIdx.x;
    if (i < n) dst[i] = (_Float16)src[i];
}

// src [R][Ccols] f32 -> dst [Ccols][R] f16
__global__ void transpose_f32_to_f16(const float* __restrict__ src, _Float16* __restrict__ dst,
                                     int R, int Ccols) {
    int i = blockIdx.x * blockDim.x + threadIdx.x;
    if (i < R * Ccols) {
        int k = i % R;
        int j = i / R;
        dst[i] = (_Float16)src[(size_t)k * Ccols + j];
    }
}

// ---------------- LDS-staged GEMM core ----------------
// Block: 256 thr = 8 waves arranged 4(M) x 2(N); block tile 128x128.
// A and B^T tiles (128 x 64 K-chunk) staged in LDS via async copies, double
// buffered: copy of chunk i+1 overlaps the 16 WMMAs of chunk i per wave.

struct GemmAcc { v8f acc[2][4]; };

template <typename Epilogue>
static __device__ __forceinline__ void gemm_tile_128x128(const _Float16* __restrict__ Ag,
                                                         const _Float16* __restrict__ Bg,
                                                         Epilogue epi) {
    __shared__ __align__(16) _Float16 atile[2][128 * LDK];   // 18 KB each
    __shared__ __align__(16) _Float16 btile[2][128 * LDK];   // 18 KB each

    const int tid  = threadIdx.x;
    const int lane = tid & 31;
    const int wave = tid >> 5;
    const int wm = wave >> 1;          // 0..3
    const int wn = wave & 1;           // 0..1

    // Copy slots: 128 rows x 64 el per tile; thread = (row = tid&127, 32-el seg).
    const int crow = tid & 127;
    const int cseg = (tid >> 7) * 32;

    auto issue_ab = [&](int buf, int k0) {
#pragma unroll
        for (int s = 0; s < 4; ++s) {
            const int el = cseg + s * 8;
            unsigned adst = (unsigned)(size_t)(const void*)&atile[buf][crow * LDK + el];
            async_copy_b128(adst, (const void*)(Ag + (size_t)crow * CQ + k0 + el));
            unsigned bdst = (unsigned)(size_t)(const void*)&btile[buf][crow * LDK + el];
            async_copy_b128(bdst, (const void*)(Bg + (size_t)crow * CQ + k0 + el));
        }
    };

    v8f acc[2][4] = {};
    issue_ab(0, 0);
    for (int k0 = 0, step = 0; k0 < CQ; k0 += 64, ++step) {
        const int buf = step & 1;
        if (k0 + 64 < CQ) {
            issue_ab(buf ^ 1, k0 + 64);                     // overlap next chunk copy
            asm volatile("s_wait_asynccnt 8" ::: "memory"); // current chunk's 8 ops done
        } else {
            asm volatile("s_wait_asynccnt 0" ::: "memory");
        }
        __syncthreads();
        const _Float16* at = &atile[buf][0];
        const _Float16* bt = &btile[buf][0];
#pragma unroll
        for (int kk = 0; kk < 64; kk += 32) {
            v16h a0 = load_frag16(at + (wm * 32) * LDK + kk, LDK, lane);
            v16h a1 = load_frag16(at + (wm * 32 + 16) * LDK + kk, LDK, lane);
#pragma unroll
            for (int j = 0; j < 4; ++j) {
                v16h bf = load_frag16(bt + (wn * 64 + j * 16) * LDK + kk, LDK, lane);
                acc[0][j] = wmma_f16(a0, bf, acc[0][j]);
                acc[1][j] = wmma_f16(a1, bf, acc[1][j]);
            }
        }
        __syncthreads();   // all waves done reading `buf` before overwrite
    }
    GemmAcc r;
#pragma unroll
    for (int i = 0; i < 2; ++i)
#pragma unroll
        for (int j = 0; j < 4; ++j) r.acc[i][j] = acc[i][j];
    epi(r, wm, wn, lane);
}

// ---------------- QKV GEMM: qkv = xh @ Wqkv + bqkv, scattered to Q/K/Vt ----------------
// Grid: (8192/128) * (3072/128) = 64*24 = 1536 blocks.

__global__ __launch_bounds__(256) void qkv_gemm(const _Float16* __restrict__ xh,
                                                const _Float16* __restrict__ WqkvT,
                                                const float* __restrict__ bqkv,
                                                _Float16* __restrict__ Qd,
                                                _Float16* __restrict__ Kd,
                                                _Float16* __restrict__ Vt) {
    const int nblk = blockIdx.x % 24;
    const int mblk = blockIdx.x / 24;
    const int row0b = mblk * 128;
    const int col0b = nblk * 128;

    gemm_tile_128x128(xh + (size_t)row0b * CQ, WqkvT + (size_t)col0b * CQ,
        [=](const GemmAcc& r, int wm, int wn, int lane) {
            const int row0 = row0b + wm * 32;
            const int col0 = col0b + wn * 64;
            // col0..col0+63 lies in exactly one Q/K/V segment (1024 % 128 == 0).
            const int seg     = col0 >> 10;    // 0=Q, 1=K, 2=V
            const int colseg0 = col0 & 1023;
#pragma unroll
            for (int j = 0; j < 4; ++j) {
                const int cs = colseg0 + j * 16 + (lane & 15);
                const int h  = cs >> 6;
                const int d  = cs & 63;
                const float bias = bqkv[seg * CQ + cs];
#pragma unroll
                for (int i = 0; i < 2; ++i) {
#pragma unroll
                    for (int rr = 0; rr < 8; ++rr) {
                        const int row  = row0 + i * 16 + rr + ((lane >> 4) << 3);
                        const int b    = row >> 11;     // / 2048
                        const int npos = row & 2047;
                        const float v = r.acc[i][j][rr] + bias;
                        if (seg == 0)
                            Qd[((size_t)(b * HQ + h) * NQ + npos) * DQ + d] = (_Float16)(v * 0.125f);
                        else if (seg == 1)
                            Kd[((size_t)(b * HQ + h) * NQ + npos) * DQ + d] = (_Float16)v;
                        else
                            Vt[((size_t)(b * HQ + h) * DQ + d) * NQ + npos] = (_Float16)v;
                    }
                }
            }
        });
}

// ---------------- fused flash attention (async double-buffered K/V in LDS) ----------------
// Grid: B*H*(N/128) = 1024 blocks of 256 thr (8 waves); each wave owns 16 queries.

__global__ __launch_bounds__(256) void flash_attn(const _Float16* __restrict__ Qd,
                                                  const _Float16* __restrict__ Kd,
                                                  const _Float16* __restrict__ Vt,
                                                  _Float16* __restrict__ attnOut) {
    __shared__ __align__(16) _Float16 ktile[2][32 * LDK];   // [key][d], padded pitch
    __shared__ __align__(16) _Float16 vtile[2][64 * LDV];   // [d][key], padded pitch
    __shared__ __align__(16) _Float16 lds_p[8][16 * LDV];   // per-wave P staging

    const int tid  = threadIdx.x;
    const int lane = tid & 31;
    const int wave = tid >> 5;
    const int qblk = blockIdx.x & 15;
    const int bh   = blockIdx.x >> 4;
    const int b    = bh >> 4;
    const int h    = bh & 15;
    const int q0   = qblk * 128 + wave * 16;

    const _Float16* Qp = Qd + ((size_t)bh * NQ + q0) * DQ;
    const _Float16* Kp = Kd + (size_t)bh * NQ * DQ;
    const _Float16* Vp = Vt + (size_t)bh * DQ * NQ;   // [D][N]

    v16h qa0 = load_frag16(Qp, DQ, lane);       // d = 0..31   (Q pre-scaled)
    v16h qa1 = load_frag16(Qp + 32, DQ, lane);  // d = 32..63

    // Per-thread 16B copy slots: K tile = 32 rows x 128 B, V tile = 64 rows x 64 B.
    const int krow = tid >> 3, kcol = (tid & 7) * 8;   // 32 x 8 threads
    const int vrow = tid >> 2, vcol = (tid & 3) * 8;   // 64 x 4 threads

    auto issue_tile = [&](int buf, int kt) {
        unsigned kdst = (unsigned)(size_t)(const void*)&ktile[buf][krow * LDK + kcol];
        async_copy_b128(kdst, (const void*)(Kp + (size_t)(kt + krow) * DQ + kcol));
        unsigned vdst = (unsigned)(size_t)(const void*)&vtile[buf][vrow * LDV + vcol];
        async_copy_b128(vdst, (const void*)(Vp + (size_t)vrow * NQ + kt + vcol));
    };

    float mrow[8], lrow[8], alpha[8];
    v8f o[4] = {};
#pragma unroll
    for (int r = 0; r < 8; ++r) { mrow[r] = -1e30f; lrow[r] = 0.f; }

    _Float16* pbuf = &lds_p[wave][0];

    issue_tile(0, 0);   // prime buffer 0 (2 async ops per wave)

    for (int kt = 0; kt < NQ; kt += 32) {
        const int buf = (kt >> 5) & 1;
        if (kt + 32 < NQ) {
            issue_tile(buf ^ 1, kt + 32);                   // overlap next tile copy
            asm volatile("s_wait_asynccnt 2" ::: "memory"); // current tile's 2 ops done
        } else {
            asm volatile("s_wait_asynccnt 0" ::: "memory");
        }
        __syncthreads();   // current tile visible to all waves

        const _Float16* kb  = &ktile[buf][0];
        const _Float16* vbt = &vtile[buf][0];

        // S tiles for keys [kt, kt+16) and [kt+16, kt+32), accumulated over D=64
        v8f s0 = {}, s1 = {};
        {
            v16h k00 = load_frag16(kb, LDK, lane);
            v16h k01 = load_frag16(kb + 32, LDK, lane);
            s0 = wmma_f16(qa0, k00, s0);
            s0 = wmma_f16(qa1, k01, s0);
            v16h k10 = load_frag16(kb + 16 * LDK, LDK, lane);
            v16h k11 = load_frag16(kb + 16 * LDK + 32, LDK, lane);
            s1 = wmma_f16(qa0, k10, s1);
            s1 = wmma_f16(qa1, k11, s1);
        }
        // Online softmax. D-layout: lane holds col=lane&15; vgpr r = row r+8*(lane>>4),
        // so rows 0-7 live in lanes 0-15, rows 8-15 in lanes 16-31: xor masks 1,2,4,8
        // reduce across the 16 key-columns without crossing halves.
#pragma unroll
        for (int r = 0; r < 8; ++r) {
            float smax = fmaxf(s0[r], s1[r]);
            smax = fmaxf(smax, __shfl_xor(smax, 1, 32));
            smax = fmaxf(smax, __shfl_xor(smax, 2, 32));
            smax = fmaxf(smax, __shfl_xor(smax, 4, 32));
            smax = fmaxf(smax, __shfl_xor(smax, 8, 32));
            const float mnew = fmaxf(mrow[r], smax);
            const float a  = __expf(mrow[r] - mnew);
            const float p0 = __expf(s0[r] - mnew);
            const float p1 = __expf(s1[r] - mnew);
            float psum = p0 + p1;
            psum += __shfl_xor(psum, 1, 32);
            psum += __shfl_xor(psum, 2, 32);
            psum += __shfl_xor(psum, 4, 32);
            psum += __shfl_xor(psum, 8, 32);
            lrow[r] = lrow[r] * a + psum;
            mrow[r] = mnew;
            alpha[r] = a;
            const int prow = r + ((lane >> 4) << 3);
            const int pcol = lane & 15;
            pbuf[prow * LDV + pcol]      = (_Float16)p0;
            pbuf[prow * LDV + 16 + pcol] = (_Float16)p1;
        }
        asm volatile("s_wait_dscnt 0" ::: "memory");
        v16h pa = load_frag16(pbuf, LDV, lane);   // P as A fragment, 16 q x 32 k
#pragma unroll
        for (int d = 0; d < 4; ++d) {
#pragma unroll
            for (int r = 0; r < 8; ++r) o[d][r] *= alpha[r];
            v16h vbf = load_frag16(vbt + d * 16 * LDV, LDV, lane);
            o[d] = wmma_f16(pa, vbf, o[d]);
        }
        __syncthreads();   // all waves done reading `buf` before it is overwritten
    }

    // normalize + write back as b n (h d), f16 for the out-projection GEMM
#pragma unroll
    for (int r = 0; r < 8; ++r) {
        const float inv = 1.0f / lrow[r];
        const int row = q0 + r + ((lane >> 4) << 3);
        const size_t base = ((size_t)b * NQ + row) * CQ + (size_t)h * DQ;
#pragma unroll
        for (int d = 0; d < 4; ++d)
            attnOut[base + d * 16 + (lane & 15)] = (_Float16)(o[d][r] * inv);
    }
}

// ---------------- output projection: out = ah @ Wout + bout (f32 out) ----------------
// Grid: (8192/128) * (1024/128) = 64*8 = 512 blocks.

__global__ __launch_bounds__(256) void out_gemm(const _Float16* __restrict__ ah,
                                                const _Float16* __restrict__ WoutT,
                                                const float* __restrict__ bout,
                                                float* __restrict__ out) {
    const int nblk = blockIdx.x & 7;
    const int mblk = blockIdx.x >> 3;
    const int row0b = mblk * 128;
    const int col0b = nblk * 128;

    gemm_tile_128x128(ah + (size_t)row0b * CQ, WoutT + (size_t)col0b * CQ,
        [=](const GemmAcc& r, int wm, int wn, int lane) {
            const int row0 = row0b + wm * 32;
            const int col0 = col0b + wn * 64;
#pragma unroll
            for (int j = 0; j < 4; ++j) {
                const int col = col0 + j * 16 + (lane & 15);
                const float bias = bout[col];
#pragma unroll
                for (int i = 0; i < 2; ++i) {
#pragma unroll
                    for (int rr = 0; rr < 8; ++rr) {
                        const int row = row0 + i * 16 + rr + ((lane >> 4) << 3);
                        out[(size_t)row * CQ + col] = r.acc[i][j][rr] + bias;
                    }
                }
            }
        });
}

// ---------------- launch ----------------

extern "C" void kernel_launch(void* const* d_in, const int* in_sizes, int n_in,
                              void* d_out, int out_size, void* d_ws, size_t ws_size,
                              hipStream_t stream) {
    (void)in_sizes; (void)n_in; (void)out_size; (void)ws_size;
    const float* x    = (const float*)d_in[0];
    const float* Wqkv = (const float*)d_in[1];
    const float* bqkv = (const float*)d_in[2];
    const float* Wout = (const float*)d_in[3];
    const float* bout = (const float*)d_in[4];
    float* out = (float*)d_out;

    char* ws = (char*)d_ws;
    size_t off = 0;
    auto carve = [&](size_t bytes) -> void* {
        void* p = ws + off;
        off += (bytes + 255) & ~(size_t)255;
        return p;
    };
    _Float16* xh    = (_Float16*)carve((size_t)BQ * NQ * CQ * 2);       // 16 MiB
    _Float16* WqkvT = (_Float16*)carve((size_t)3 * CQ * CQ * 2);        // 6 MiB
    _Float16* WoutT = (_Float16*)carve((size_t)CQ * CQ * 2);            // 2 MiB
    _Float16* Qd    = (_Float16*)carve((size_t)BQ * HQ * NQ * DQ * 2);  // 16 MiB
    _Float16* Kd    = (_Float16*)carve((size_t)BQ * HQ * NQ * DQ * 2);  // 16 MiB
    _Float16* Vt    = (_Float16*)carve((size_t)BQ * HQ * NQ * DQ * 2);  // 16 MiB
    _Float16* ah    = (_Float16*)carve((size_t)BQ * NQ * CQ * 2);       // 16 MiB

    const int nx = BQ * NQ * CQ;
    cvt_f32_to_f16<<<(nx + 255) / 256, 256, 0, stream>>>(x, xh, nx);
    const int nw = 3 * CQ * CQ;
    transpose_f32_to_f16<<<(nw + 255) / 256, 256, 0, stream>>>(Wqkv, WqkvT, CQ, 3 * CQ);
    const int nwo = CQ * CQ;
    transpose_f32_to_f16<<<(nwo + 255) / 256, 256, 0, stream>>>(Wout, WoutT, CQ, CQ);

    qkv_gemm<<<64 * 24, 256, 0, stream>>>(xh, WqkvT, bqkv, Qd, Kd, Vt);
    flash_attn<<<BQ * HQ * (NQ / 128), 256, 0, stream>>>(Qd, Kd, Vt, ah);
    out_gemm<<<64 * 8, 256, 0, stream>>>(ah, WoutT, bout, out);
}
